// MultiScaleHGPSL_9208409883079
// MI455X (gfx1250) — compile-verified, matrix operand order checked by reading, not run on hardware
//
#include <hip/hip_runtime.h>
#include <hip/hip_bf16.h>
#include <stdint.h>
#include <stddef.h>

typedef unsigned short ushort_t;
typedef __attribute__((ext_vector_type(16))) __bf16 v16bf;
typedef __attribute__((ext_vector_type(8)))  float  v8f;

#define NT_  65536
#define G_   64
#define NPG_ 1024
#define OUT_ 128
#define NEG_INF_ (-3.402823466e38f)

// ---------------- device helpers ----------------

__device__ __forceinline__ unsigned short f32_bf16(float f) {
  union { float f; unsigned u; } v; v.f = f;
  unsigned u = v.u;
  unsigned r = 0x7fffu + ((u >> 16) & 1u);
  return (unsigned short)((u + r) >> 16);
}

__device__ __forceinline__ void atomic_max_f32(float* addr, float val) {
  unsigned* ua = (unsigned*)addr;
  unsigned old = *ua;
  while (__uint_as_float(old) < val) {
    unsigned assumed = old;
    old = atomicCAS(ua, assumed, __float_as_uint(val));
    if (old == assumed) break;
  }
}

// async global->LDS copy of 16 bytes (CDNA5): ldsOff is the LDS byte address,
// gp the 64-bit global address. Tracked by ASYNCcnt.
__device__ __forceinline__ void async_copy_b128(unsigned ldsOff, const void* gp) {
  asm volatile("global_load_async_to_lds_b128 %0, %1, off"
               :: "v"(ldsOff), "v"(gp) : "memory");
}
__device__ __forceinline__ void wait_asynccnt0() {
  asm volatile("s_wait_asynccnt 0x0" ::: "memory");
}
__device__ __forceinline__ unsigned lds_addr_of(const void* p) {
  return (unsigned)(unsigned long long)p;  // low 32 bits of LDS-aperture address
}

// ---------------- generic fill / pack kernels ----------------

__global__ __launch_bounds__(256) void k_fill(float* p, float v, unsigned n) {
  unsigned gid = blockIdx.x * 256u + threadIdx.x;
  if (gid < n) p[gid] = v;
}

// dst[m, k] = k < cols ? bf16(src[m, k]) : 0      dst row stride Kp
__global__ __launch_bounds__(256) void k_pack(const float* __restrict__ src,
                                              ushort_t* __restrict__ dst,
                                              unsigned n, int cols, int Kp) {
  unsigned gid = blockIdx.x * 256u + threadIdx.x;
  if (gid >= n) return;
  unsigned m = gid / (unsigned)Kp, k = gid % (unsigned)Kp;
  dst[gid] = ((int)k < cols) ? f32_bf16(src[(size_t)m * cols + k]) : (ushort_t)0;
}

// dst[m, k] = bf16(a[m,k] + b[m,k])  (GIN: h + agg)
__global__ __launch_bounds__(256) void k_pack_add(const float* __restrict__ a,
                                                  const float* __restrict__ b,
                                                  ushort_t* __restrict__ dst,
                                                  unsigned n, int cols, int Kp) {
  unsigned gid = blockIdx.x * 256u + threadIdx.x;
  if (gid >= n) return;
  unsigned m = gid / (unsigned)Kp, k = gid % (unsigned)Kp;
  float v = 0.0f;
  if ((int)k < cols) v = a[(size_t)m * cols + k] + b[(size_t)m * cols + k];
  dst[gid] = ((int)k < cols) ? f32_bf16(v) : (ushort_t)0;
}

// dst[m, k] = concat(a[m,:cols], b[m,:cols]) in bf16 (MR)
__global__ __launch_bounds__(256) void k_pack2(const float* __restrict__ a,
                                               const float* __restrict__ b,
                                               ushort_t* __restrict__ dst,
                                               unsigned n, int cols, int Kp) {
  unsigned gid = blockIdx.x * 256u + threadIdx.x;
  if (gid >= n) return;
  unsigned m = gid / (unsigned)Kp, k = gid % (unsigned)Kp;
  ushort_t out = 0;
  if ((int)k < cols)           out = f32_bf16(a[(size_t)m * cols + k]);
  else if ((int)k < 2 * cols)  out = f32_bf16(b[(size_t)m * cols + (k - cols)]);
  dst[gid] = out;
}

// W [K, Nout] f32 -> Wb [Nout, Kp] bf16 (transposed, zero-padded)
__global__ __launch_bounds__(256) void k_cvt_w(const float* __restrict__ W,
                                               ushort_t* __restrict__ Wb,
                                               unsigned n, int K, int Kp, int Nout) {
  unsigned gid = blockIdx.x * 256u + threadIdx.x;
  if (gid >= n) return;
  unsigned nn = gid / (unsigned)Kp, k = gid % (unsigned)Kp;
  Wb[gid] = ((int)k < K) ? f32_bf16(W[(size_t)k * Nout + nn]) : (ushort_t)0;
}

// ---------------- WMMA GEMM: C[M,N] = A[M,Kp](bf16) x Bt[N,Kp](bf16)^T + bias ----------------
// block = 256 threads (8 waves); block tile 128x128; wave w -> rows [w*16, w*16+16)
// Double-buffered LDS, filled with global_load_async_to_lds_b128 (ASYNCcnt) so the
// DMA of tile k+1 overlaps the WMMA issue of tile k.
// LDS rows padded to 40 bf16 (80B) to avoid ds bank conflicts on fragment loads.

__global__ __launch_bounds__(256) void k_wmma_gemm(const ushort_t* __restrict__ A,
                                                   const ushort_t* __restrict__ Bt,
                                                   const float* __restrict__ bias,
                                                   float* __restrict__ C,
                                                   int Kp, int Nc) {
  __shared__ ushort_t As[2][128 * 40];
  __shared__ ushort_t Bs[2][128 * 40];
  const int tid  = threadIdx.x;
  const int lane = tid & 31;
  const int wv   = tid >> 5;
  const int half = lane >> 4;
  const int l16  = lane & 15;
  const int m0 = blockIdx.y * 128;
  const int n0 = blockIdx.x * 128;

  v8f acc[8];
#pragma unroll
  for (int t = 0; t < 8; ++t)
#pragma unroll
    for (int r = 0; r < 8; ++r) acc[t][r] = 0.0f;

  // issue async stage of one 128x32 K-tile of A and Bt into LDS buffer pb
  auto stage = [&](int k0, int pb) {
#pragma unroll
    for (int i = 0; i < 2; ++i) {
      int idx = tid + i * 256;
      int row = idx >> 2, seg = idx & 3;
      async_copy_b128(lds_addr_of(&As[pb][row * 40 + seg * 8]),
                      A + (size_t)(m0 + row) * Kp + k0 + seg * 8);
      async_copy_b128(lds_addr_of(&Bs[pb][row * 40 + seg * 8]),
                      Bt + (size_t)(n0 + row) * Kp + k0 + seg * 8);
    }
  };

  int pb = 0;
  stage(0, 0);
  wait_asynccnt0();
  __syncthreads();

  for (int k0 = 0; k0 < Kp; k0 += 32) {
    const int nk = k0 + 32;
    if (nk < Kp) stage(nk, pb ^ 1);           // DMA next tile while computing
    if (nk + 32 < Kp)                         // speculative prefetch of tile after next
      __builtin_prefetch(A + (size_t)(m0 + (tid >> 1)) * Kp + nk + 32, 0, 0);

    const ushort_t* Asb = As[pb];
    const ushort_t* Bsb = Bs[pb];

    // A fragment: row M = wv*16 + l16; K pairs per the CDNA5 16-bit A layout
    union FragU { v16bf v; unsigned u[8]; };
    FragU aF;
    const int arow = wv * 16 + l16;
#pragma unroll
    for (int p = 0; p < 8; ++p) {
      int kk = (p < 4) ? (half * 8 + p * 2) : (16 + half * 8 + (p - 4) * 2);
      aF.u[p] = *(const unsigned*)&Asb[arow * 40 + kk];
    }
    // preload all 8 B fragments, then issue WMMAs back-to-back
    FragU bF[8];
#pragma unroll
    for (int t = 0; t < 8; ++t) {
      const int brow = t * 16 + l16;
#pragma unroll
      for (int p = 0; p < 8; ++p) {
        int kk = (p < 4) ? (half * 8 + p * 2) : (16 + half * 8 + (p - 4) * 2);
        bF[t].u[p] = *(const unsigned*)&Bsb[brow * 40 + kk];
      }
    }
#pragma unroll
    for (int t = 0; t < 8; ++t)
      acc[t] = __builtin_amdgcn_wmma_f32_16x16x32_bf16(
          false, aF.v, false, bF[t].v, (short)0, acc[t], false, false);

    if (nk < Kp) wait_asynccnt0();            // next tile fully in LDS
    __syncthreads();                          // all waves done reading buf pb
    pb ^= 1;
  }

#pragma unroll
  for (int t = 0; t < 8; ++t) {
    const int n = n0 + t * 16 + l16;
    float b = bias ? bias[n] : 0.0f;
#pragma unroll
    for (int r = 0; r < 8; ++r) {
      int m = m0 + wv * 16 + half * 8 + r;
      C[(size_t)m * Nc + n] = acc[t][r] + b;
    }
  }
}

// ---------------- layernorm + ELU (wave per row, wave32) ----------------

__global__ __launch_bounds__(256) void k_ln_elu(float* __restrict__ h, int d) {
  int row  = blockIdx.x * 8 + (threadIdx.x >> 5);
  int lane = threadIdx.x & 31;
  float* p = h + (size_t)row * d;
  float s = 0.0f;
  for (int i = lane; i < d; i += 32) s += p[i];
#pragma unroll
  for (int o = 16; o > 0; o >>= 1) s += __shfl_down(s, o, 32);
  float mean = __shfl(s, 0, 32) / (float)d;
  float v = 0.0f;
  for (int i = lane; i < d; i += 32) { float x = p[i] - mean; v += x * x; }
#pragma unroll
  for (int o = 16; o > 0; o >>= 1) v += __shfl_down(v, o, 32);
  float inv = rsqrtf(__shfl(v, 0, 32) / (float)d + 1e-5f);
  for (int i = lane; i < d; i += 32) {
    float x = (p[i] - mean) * inv;
    p[i] = x > 0.0f ? x : (expf(x) - 1.0f);
  }
}

// ---------------- edge kernels ----------------

// EdgeConv: out[dst] = max(msg), msg = t[src]-t[dst]+phi[dst]
__global__ __launch_bounds__(256) void k_edge_msg_max(const float* __restrict__ t,
                                                      const float* __restrict__ phi,
                                                      const int* __restrict__ src,
                                                      const int* __restrict__ dst,
                                                      float* __restrict__ out,
                                                      unsigned n, int d) {
  unsigned gid = blockIdx.x * 256u + threadIdx.x;
  if (gid >= n) return;
  unsigned e = gid / (unsigned)d, c = gid % (unsigned)d;
  int s = src[e], v = dst[e];
  float msg = t[(size_t)s * d + c] - t[(size_t)v * d + c] + phi[(size_t)v * d + c];
  atomic_max_f32(out + (size_t)v * d + c, msg);
}

// GIN: out[dst] = max(h[src])
__global__ __launch_bounds__(256) void k_seg_max_src(const float* __restrict__ h,
                                                     const int* __restrict__ src,
                                                     const int* __restrict__ dst,
                                                     float* __restrict__ out,
                                                     unsigned n, int d) {
  unsigned gid = blockIdx.x * 256u + threadIdx.x;
  if (gid >= n) return;
  unsigned e = gid / (unsigned)d, c = gid % (unsigned)d;
  int s = src[e], v = dst[e];
  atomic_max_f32(out + (size_t)v * d + c, h[(size_t)s * d + c]);
}

// MR: out[dst] = max(h[src]-h[dst])
__global__ __launch_bounds__(256) void k_seg_max_diff(const float* __restrict__ h,
                                                      const int* __restrict__ src,
                                                      const int* __restrict__ dst,
                                                      float* __restrict__ out,
                                                      unsigned n, int d) {
  unsigned gid = blockIdx.x * 256u + threadIdx.x;
  if (gid >= n) return;
  unsigned e = gid / (unsigned)d, c = gid % (unsigned)d;
  int s = src[e], v = dst[e];
  atomic_max_f32(out + (size_t)v * d + c,
                 h[(size_t)s * d + c] - h[(size_t)v * d + c]);
}

// GAT attention scores: el[n,h] = sum_i ft[n,h,i]*al[h,i]; er similarly
__global__ __launch_bounds__(256) void k_gat_scores(const float* __restrict__ ft,
                                                    const float* __restrict__ al,
                                                    const float* __restrict__ ar,
                                                    float* __restrict__ el,
                                                    float* __restrict__ er,
                                                    unsigned n, int H, int D) {
  unsigned gid = blockIdx.x * 256u + threadIdx.x;
  if (gid >= n) return;
  unsigned node = gid / (unsigned)H, h = gid % (unsigned)H;
  const float* f = ft + (size_t)node * H * D + (size_t)h * D;
  float a = 0.0f, b = 0.0f;
  for (int i = 0; i < D; ++i) { a += f[i] * al[h * D + i]; b += f[i] * ar[h * D + i]; }
  el[gid] = a; er[gid] = b;
}

__global__ __launch_bounds__(256) void k_gat_edge_max(const float* __restrict__ el,
                                                      const float* __restrict__ er,
                                                      const int* __restrict__ src,
                                                      const int* __restrict__ dst,
                                                      float* __restrict__ m,
                                                      unsigned n, int H) {
  unsigned gid = blockIdx.x * 256u + threadIdx.x;
  if (gid >= n) return;
  unsigned e = gid / (unsigned)H, h = gid % (unsigned)H;
  int s = src[e], v = dst[e];
  float ev = el[(size_t)s * H + h] + er[(size_t)v * H + h];
  ev = ev > 0.0f ? ev : 0.2f * ev;
  atomic_max_f32(m + (size_t)v * H + h, ev);
}

__global__ __launch_bounds__(256) void k_gat_edge_den(const float* __restrict__ el,
                                                      const float* __restrict__ er,
                                                      const float* __restrict__ m,
                                                      const int* __restrict__ src,
                                                      const int* __restrict__ dst,
                                                      float* __restrict__ den,
                                                      unsigned n, int H) {
  unsigned gid = blockIdx.x * 256u + threadIdx.x;
  if (gid >= n) return;
  unsigned e = gid / (unsigned)H, h = gid % (unsigned)H;
  int s = src[e], v = dst[e];
  float ev = el[(size_t)s * H + h] + er[(size_t)v * H + h];
  ev = ev > 0.0f ? ev : 0.2f * ev;
  atomicAdd(den + (size_t)v * H + h, expf(ev - m[(size_t)v * H + h]));
}

__global__ __launch_bounds__(256) void k_gat_aggregate(const float* __restrict__ ft,
                                                       const float* __restrict__ el,
                                                       const float* __restrict__ er,
                                                       const float* __restrict__ m,
                                                       const float* __restrict__ den,
                                                       const int* __restrict__ src,
                                                       const int* __restrict__ dst,
                                                       float* __restrict__ out,
                                                       unsigned n, int H, int D) {
  unsigned gid = blockIdx.x * 256u + threadIdx.x;
  if (gid >= n) return;
  unsigned e = gid / (unsigned)D, c = gid % (unsigned)D;
  int s = src[e], v = dst[e];
  for (int h = 0; h < H; ++h) {
    float ev = el[(size_t)s * H + h] + er[(size_t)v * H + h];
    ev = ev > 0.0f ? ev : 0.2f * ev;
    float alpha = expf(ev - m[(size_t)v * H + h]) / den[(size_t)v * H + h];
    atomicAdd(out + (size_t)v * H * D + (size_t)h * D + c,
              alpha * ft[(size_t)s * H * D + (size_t)h * D + c]);
  }
}

// mean-pool per graph into d_out[G, 512] at column offset `base`
__global__ __launch_bounds__(256) void k_pool_add(const float* __restrict__ h,
                                                  const int* __restrict__ gidarr,
                                                  float* __restrict__ out,
                                                  unsigned n, int d, int base) {
  unsigned gid = blockIdx.x * 256u + threadIdx.x;
  if (gid >= n) return;
  unsigned node = gid / (unsigned)d, c = gid % (unsigned)d;
  atomicAdd(out + (size_t)gidarr[node] * 512 + base + c,
            h[(size_t)node * d + c] * (1.0f / (float)NPG_));
}

// ---------------- host orchestration ----------------

static inline unsigned cdiv_u(unsigned a, unsigned b) { return (a + b - 1u) / b; }
static inline int pad32(int k) { return (k + 31) & ~31; }

extern "C" void kernel_launch(void* const* d_in, const int* in_sizes, int n_in,
                              void* d_out, int out_size, void* d_ws, size_t ws_size,
                              hipStream_t stream) {
  (void)n_in; (void)out_size; (void)ws_size;
  const float* feats = (const float*)d_in[0];
  const int* esrc = (const int*)d_in[1];
  const int* edst = (const int*)d_in[2];
  const int* ngid = (const int*)d_in[3];
  const unsigned E = (unsigned)in_sizes[1];

  const float* Wt_[3] = {(const float*)d_in[4],  (const float*)d_in[5],  (const float*)d_in[6]};
  const float* Wp_[3] = {(const float*)d_in[7],  (const float*)d_in[8],  (const float*)d_in[9]};
  const float* bt_[3] = {(const float*)d_in[10], (const float*)d_in[11], (const float*)d_in[12]};
  const float* bp_[3] = {(const float*)d_in[13], (const float*)d_in[14], (const float*)d_in[15]};
  const float* gW_[3] = {(const float*)d_in[16], (const float*)d_in[17], (const float*)d_in[18]};
  const float* gal_[3]= {(const float*)d_in[19], (const float*)d_in[20], (const float*)d_in[21]};
  const float* gar_[3]= {(const float*)d_in[22], (const float*)d_in[23], (const float*)d_in[24]};
  const float* iW_[3] = {(const float*)d_in[25], (const float*)d_in[26], (const float*)d_in[27]};
  const float* ib_[3] = {(const float*)d_in[28], (const float*)d_in[29], (const float*)d_in[30]};
  const float* mW_[3] = {(const float*)d_in[31], (const float*)d_in[32], (const float*)d_in[33]};
  const float* mb_[3] = {(const float*)d_in[34], (const float*)d_in[35], (const float*)d_in[36]};

  // workspace carve-out
  size_t off = 0;
  auto alloc = [&](size_t bytes) -> void* {
    void* p = (char*)d_ws + off;
    off += (bytes + 255) & ~(size_t)255;
    return p;
  };
  ushort_t* Ab  = (ushort_t*)alloc((size_t)NT_ * 576 * 2);
  ushort_t* Wb  = (ushort_t*)alloc((size_t)576 * 256 * 2);
  float* buf0 = (float*)alloc((size_t)NT_ * 256 * 4);
  float* buf1 = (float*)alloc((size_t)NT_ * 256 * 4);
  float* buf2 = (float*)alloc((size_t)NT_ * 256 * 4);
  float* buf3 = (float*)alloc((size_t)NT_ * 256 * 4);
  float* elb  = (float*)alloc((size_t)NT_ * 2 * 4);
  float* erb  = (float*)alloc((size_t)NT_ * 2 * 4);
  float* mb   = (float*)alloc((size_t)NT_ * 2 * 4);
  float* denb = (float*)alloc((size_t)NT_ * 2 * 4);

  float* outp = (float*)d_out;

  auto fill = [&](float* p, float v, unsigned n) {
    k_fill<<<cdiv_u(n, 256), 256, 0, stream>>>(p, v, n);
  };
  auto gemm = [&](const float* W, const float* bias, float* C, int K, int Kp, int Nout) {
    unsigned nw = (unsigned)Nout * Kp;
    k_cvt_w<<<cdiv_u(nw, 256), 256, 0, stream>>>(W, Wb, nw, K, Kp, Nout);
    dim3 grid(Nout / 128, NT_ / 128);
    k_wmma_gemm<<<grid, 256, 0, stream>>>(Ab, Wb, bias, C, Kp, Nout);
  };
  auto lnelu = [&](float* h, int d) {
    k_ln_elu<<<NT_ / 8, 256, 0, stream>>>(h, d);
  };

  fill(outp, 0.0f, G_ * 512);

  // =============== EdgeConv backbone ===============
  {
    const float* h = feats; int hc = 281;
    float* hbufs[2] = {buf2, buf3};
    for (int l = 0; l < 3; ++l) {
      int Kp = pad32(hc);
      unsigned np = (unsigned)NT_ * Kp;
      k_pack<<<cdiv_u(np, 256), 256, 0, stream>>>(h, Ab, np, hc, Kp);
      gemm(Wt_[l], bt_[l], buf0, hc, Kp, 128);   // t
      gemm(Wp_[l], bp_[l], buf1, hc, Kp, 128);   // phi
      float* hout = hbufs[l & 1];
      fill(hout, NEG_INF_, NT_ * 128);
      unsigned ne = E * 128u;
      k_edge_msg_max<<<cdiv_u(ne, 256), 256, 0, stream>>>(buf0, buf1, esrc, edst,
                                                          hout, ne, 128);
      lnelu(hout, 128);
      h = hout; hc = 128;
    }
    unsigned nn = (unsigned)NT_ * 128;
    k_pool_add<<<cdiv_u(nn, 256), 256, 0, stream>>>((const float*)h, ngid, outp, nn, 128, 0);
  }

  // =============== GAT backbone ===============
  {
    const int HH[3] = {2, 2, 1};
    const float* h = feats; int hc = 281;
    float* obufs[3] = {buf1, buf2, buf1};
    for (int l = 0; l < 3; ++l) {
      int H = HH[l], D = 128, HD = H * D;
      int Kp = pad32(hc);
      unsigned np = (unsigned)NT_ * Kp;
      k_pack<<<cdiv_u(np, 256), 256, 0, stream>>>(h, Ab, np, hc, Kp);
      gemm(gW_[l], nullptr, buf0, hc, Kp, HD);   // ft
      unsigned ns = (unsigned)NT_ * H;
      k_gat_scores<<<cdiv_u(ns, 256), 256, 0, stream>>>(buf0, gal_[l], gar_[l],
                                                        elb, erb, ns, H, D);
      fill(mb, NEG_INF_, ns);
      unsigned neh = E * (unsigned)H;
      k_gat_edge_max<<<cdiv_u(neh, 256), 256, 0, stream>>>(elb, erb, esrc, edst, mb, neh, H);
      fill(denb, 0.0f, ns);
      k_gat_edge_den<<<cdiv_u(neh, 256), 256, 0, stream>>>(elb, erb, mb, esrc, edst,
                                                           denb, neh, H);
      float* hout = obufs[l];
      fill(hout, 0.0f, (unsigned)NT_ * HD);
      unsigned ned = E * 128u;
      k_gat_aggregate<<<cdiv_u(ned, 256), 256, 0, stream>>>(buf0, elb, erb, mb, denb,
                                                            esrc, edst, hout, ned, H, D);
      lnelu(hout, HD);
      h = hout; hc = HD;
    }
    unsigned nn = (unsigned)NT_ * 128;
    k_pool_add<<<cdiv_u(nn, 256), 256, 0, stream>>>((const float*)h, ngid, outp, nn, 128, 128);
  }

  // =============== GIN backbone ===============
  {
    const float* h = feats; int hc = 281;
    float* obufs[3] = {buf1, buf2, buf1};
    for (int l = 0; l < 3; ++l) {
      fill(buf0, NEG_INF_, (unsigned)NT_ * hc);
      unsigned ne = E * (unsigned)hc;
      k_seg_max_src<<<cdiv_u(ne, 256), 256, 0, stream>>>(h, esrc, edst, buf0, ne, hc);
      int Kp = pad32(hc);
      unsigned np = (unsigned)NT_ * Kp;
      k_pack_add<<<cdiv_u(np, 256), 256, 0, stream>>>(h, buf0, Ab, np, hc, Kp);
      float* hout = obufs[l];
      gemm(iW_[l], ib_[l], hout, hc, Kp, 128);
      lnelu(hout, 128);
      h = hout; hc = 128;
    }
    unsigned nn = (unsigned)NT_ * 128;
    k_pool_add<<<cdiv_u(nn, 256), 256, 0, stream>>>((const float*)h, ngid, outp, nn, 128, 256);
  }

  // =============== Max-Relative backbone ===============
  {
    const float* h = feats; int hc = 281;
    float* obufs[3] = {buf1, buf2, buf1};
    for (int l = 0; l < 3; ++l) {
      fill(buf0, NEG_INF_, (unsigned)NT_ * hc);
      unsigned ne = E * (unsigned)hc;
      k_seg_max_diff<<<cdiv_u(ne, 256), 256, 0, stream>>>(h, esrc, edst, buf0, ne, hc);
      int K = 2 * hc, Kp = pad32(K);
      unsigned np = (unsigned)NT_ * Kp;
      k_pack2<<<cdiv_u(np, 256), 256, 0, stream>>>(h, buf0, Ab, np, hc, Kp);
      float* hout = obufs[l];
      gemm(mW_[l], mb_[l], hout, K, Kp, 128);
      lnelu(hout, 128);
      h = hout; hc = 128;
    }
    unsigned nn = (unsigned)NT_ * 128;
    k_pool_add<<<cdiv_u(nn, 256), 256, 0, stream>>>((const float*)h, ngid, outp, nn, 128, 384);
  }
}
